// ExactLlamaAttention_73693048865147
// MI455X (gfx1250) — matrix operand order repaired and checked by reading, not compiled, not run on hardware
//
#include <hip/hip_runtime.h>
#include <hip/hip_bf16.h>
#include <math.h>

#define S_LEN 2048
#define D_MODEL 4096
#define N_HEADS 32
#define N_KVH 8
#define HEAD_DIM 128

typedef __bf16 bf16_t;
typedef __attribute__((ext_vector_type(16))) bf16_t v16bf;
typedef __attribute__((ext_vector_type(8)))  float  v8f;
typedef __attribute__((ext_vector_type(4)))  int    v4i;

union Frag {
    v16bf v;
    uint4 q[2];
};

__device__ __forceinline__ unsigned short f2bf(float f) {
    unsigned u = __float_as_uint(f);
    unsigned r = u + 0x7FFFu + ((u >> 16) & 1u);
    return (unsigned short)(r >> 16);
}

// CDNA5 async global->LDS copy (ASYNCcnt-tracked), guarded so compile never breaks.
#if defined(__gfx1250__) && __has_builtin(__builtin_amdgcn_global_load_async_to_lds_b128)
#define USE_ASYNC_LDS 1
#define ASYNC_CP(gp, lp)                                                  \
    __builtin_amdgcn_global_load_async_to_lds_b128(                       \
        (__attribute__((address_space(1))) v4i*)(gp),                     \
        (__attribute__((address_space(3))) v4i*)(lp), 0, 0)
#if __has_builtin(__builtin_amdgcn_s_wait_asynccnt)
#define ASYNC_WAIT() __builtin_amdgcn_s_wait_asynccnt(0)
#else
#define ASYNC_WAIT() asm volatile("s_wait_asynccnt 0x0" ::: "memory")
#endif
#else
#define USE_ASYNC_LDS 0
#define ASYNC_CP(gp, lp) (*(uint4*)(lp) = *(const uint4*)(gp))
#define ASYNC_WAIT()
#endif

// ---------------------------------------------------------------------------
// fp32 -> bf16 convert (grid-stride)
// ---------------------------------------------------------------------------
__global__ void k_cvt_bf16(const float* __restrict__ src,
                           unsigned short* __restrict__ dst, int n) {
    for (int i = blockIdx.x * blockDim.x + threadIdx.x; i < n;
         i += gridDim.x * blockDim.x) {
        dst[i] = f2bf(src[i]);
    }
}

// ---------------------------------------------------------------------------
// fp32 (K x N) -> bf16 transposed (N x K), LDS-tiled so both sides coalesce.
// ---------------------------------------------------------------------------
__global__ __launch_bounds__(256)
void k_cvt_tr(const float* __restrict__ src, unsigned short* __restrict__ dst,
              int K, int N) {
    __shared__ float t[32][33];
    const int tx = threadIdx.x & 31;
    const int ty = threadIdx.x >> 5;          // 0..7
    const int kb = blockIdx.y * 32;
    const int nb = blockIdx.x * 32;
#pragma unroll
    for (int r = 0; r < 4; ++r)
        t[ty + r * 8][tx] = src[(size_t)(kb + ty + r * 8) * N + nb + tx];
    __syncthreads();
#pragma unroll
    for (int r = 0; r < 4; ++r)
        dst[(size_t)(nb + ty + r * 8) * K + kb + tx] = f2bf(t[tx][ty + r * 8]);
}

// ---------------------------------------------------------------------------
// bf16 GEMM: C(MxN, f32) = A(MxK, bf16 rm) @ Bt(NxK, bf16 rm, pre-transposed)
// 128x128 block tile, 8 waves in a 4(m) x 2(n) grid, 32x64 per wave.
// Double-buffered LDS staging via async global->LDS copies.
// ---------------------------------------------------------------------------
__global__ __launch_bounds__(256)
void k_gemm_bf16(const unsigned short* __restrict__ A,
                 const unsigned short* __restrict__ Bt,
                 float* __restrict__ C, int M, int N, int K) {
    __shared__ unsigned short As[2][128][48];   // [buf][m][k], 96B row stride
    __shared__ unsigned short Bs[2][128][48];   // [buf][n][k]

    const int tid  = threadIdx.x;
    const int lane = tid & 31;
    const int wave = tid >> 5;
    const int lo   = lane & 15;
    const int hi   = lane >> 4;
    const int wm   = wave >> 1;        // 0..3
    const int wn   = wave & 1;         // 0..1
    const int mb   = blockIdx.y * 128;
    const int nb   = blockIdx.x * 128;

    v8f acc[2][4];
#pragma unroll
    for (int t = 0; t < 2; ++t)
#pragma unroll
        for (int u = 0; u < 4; ++u)
#pragma unroll
            for (int j = 0; j < 8; ++j) acc[t][u][j] = 0.0f;

    auto stage = [&](int kt, unsigned short (*as)[48], unsigned short (*bs)[48]) {
#pragma unroll
        for (int r = 0; r < 2; ++r) {
            int i   = tid + r * 256;           // 0..511
            int row = i >> 2;                  // 0..127
            int ch  = i & 3;                   // 0..3
            const unsigned short* gp = A + (size_t)(mb + row) * K + kt * 32 + ch * 8;
            ASYNC_CP(gp, &as[row][ch * 8]);
        }
#pragma unroll
        for (int r = 0; r < 2; ++r) {
            int i   = tid + r * 256;
            int row = i >> 2;
            int ch  = i & 3;
            const unsigned short* gp = Bt + (size_t)(nb + row) * K + kt * 32 + ch * 8;
            ASYNC_CP(gp, &bs[row][ch * 8]);
        }
    };

    const int kblocks = K >> 5;

    stage(0, As[0], Bs[0]);
    ASYNC_WAIT();
    __syncthreads();

    for (int kt = 0; kt < kblocks; ++kt) {
        const int cur = kt & 1;
        if (kt + 1 < kblocks) stage(kt + 1, As[cur ^ 1], Bs[cur ^ 1]);

        Frag am[2], bn[4];
#pragma unroll
        for (int t = 0; t < 2; ++t) {
            const unsigned short* ap = &As[cur][wm * 32 + t * 16 + lo][hi * 8];
            am[t].q[0] = *(const uint4*)ap;
            am[t].q[1] = *(const uint4*)(ap + 16);
        }
#pragma unroll
        for (int u = 0; u < 4; ++u) {
            const unsigned short* bp = &Bs[cur][wn * 64 + u * 16 + lo][hi * 8];
            bn[u].q[0] = *(const uint4*)bp;
            bn[u].q[1] = *(const uint4*)(bp + 16);
        }
#pragma unroll
        for (int t = 0; t < 2; ++t)
#pragma unroll
            for (int u = 0; u < 4; ++u)
                acc[t][u] = __builtin_amdgcn_wmma_f32_16x16x32_bf16(
                    false, am[t].v, false, bn[u].v, (short)0, acc[t][u],
                    false, false);

        ASYNC_WAIT();
        __syncthreads();
    }

#pragma unroll
    for (int t = 0; t < 2; ++t)
#pragma unroll
        for (int u = 0; u < 4; ++u)
#pragma unroll
            for (int v = 0; v < 8; ++v) {
                int row = mb + wm * 32 + t * 16 + v + 8 * hi;
                int col = nb + wn * 64 + u * 16 + lo;
                C[(size_t)row * N + col] = acc[t][u][v];
            }
}

// ---------------------------------------------------------------------------
// RoPE + repack: X (S, nH, 128) f32 -> out (nH, S, 128) bf16, q-scale folded.
// ---------------------------------------------------------------------------
__global__ void k_rope_pack(const float* __restrict__ X,
                            const int* __restrict__ pos,
                            unsigned short* __restrict__ out,
                            int nH, float scale) {
    int idx = blockIdx.x * blockDim.x + threadIdx.x;
    int total = S_LEN * nH * 64;
    if (idx >= total) return;
    int d = idx & 63;
    int h = (idx >> 6) % nH;
    int s = idx / (64 * nH);
    const float* xp = X + ((size_t)s * nH + h) * HEAD_DIM;
    float x1 = xp[d];
    float x2 = xp[d + 64];
    float p  = (float)pos[s];
    float inv = __powf(1.0e6f, -(float)d * (1.0f / 64.0f));
    float t = p * inv;
    float sn, cs;
    __sincosf(t, &sn, &cs);
    unsigned short* op = out + ((size_t)h * S_LEN + s) * HEAD_DIM;
    op[d]      = f2bf((x1 * cs - x2 * sn) * scale);
    op[d + 64] = f2bf((x1 * sn + x2 * cs) * scale);
}

// V (S, KVH, 128) f32 -> Vt (KVH, 128, S) bf16 (transposed for PV B-frags)
__global__ void k_pack_vT(const float* __restrict__ X,
                          unsigned short* __restrict__ out) {
    int idx = blockIdx.x * blockDim.x + threadIdx.x;
    int total = S_LEN * N_KVH * HEAD_DIM;
    if (idx >= total) return;
    int d = idx & 127;
    int h = (idx >> 7) % N_KVH;
    int s = idx / (HEAD_DIM * N_KVH);
    out[((size_t)h * HEAD_DIM + d) * S_LEN + s] =
        f2bf(X[((size_t)s * N_KVH + h) * HEAD_DIM + d]);
}

// ---------------------------------------------------------------------------
// Flash attention, one wave per (head, 16-row q tile).
// Q: (H, S, 128) bf16 (pre-scaled); K: (KVH, S, 128) bf16; Vt: (KVH, 128, S)
// ctx out: (S, H*128) bf16
// ---------------------------------------------------------------------------
__global__ __launch_bounds__(32)
void k_attn(const unsigned short* __restrict__ Q,
            const unsigned short* __restrict__ Kt,
            const unsigned short* __restrict__ Vt,
            const float* __restrict__ amask,
            unsigned short* __restrict__ ctx) {
    const int h    = blockIdx.y;
    const int qt   = blockIdx.x;
    const int lane = threadIdx.x;
    const int lo   = lane & 15;
    const int hi   = lane >> 4;
    const int g    = h >> 2;          // n_rep = H / KVH = 4
    const int q0   = qt * 16;

    const unsigned short* Qh = Q  + (size_t)h * S_LEN * HEAD_DIM;
    const unsigned short* Kh = Kt + (size_t)g * S_LEN * HEAD_DIM;
    const unsigned short* Vh = Vt + (size_t)g * HEAD_DIM * S_LEN;

    __shared__ unsigned short Pl[16][32];

    Frag qa[4];
#pragma unroll
    for (int c = 0; c < 4; ++c) {
        const unsigned short* qp = Qh + (size_t)(q0 + lo) * HEAD_DIM + c * 32 + hi * 8;
        qa[c].q[0] = *(const uint4*)qp;
        qa[c].q[1] = *(const uint4*)(qp + 16);
    }

    v8f O[8];
#pragma unroll
    for (int i = 0; i < 8; ++i)
#pragma unroll
        for (int j = 0; j < 8; ++j) O[i][j] = 0.0f;
    float rmax[8], rsum[8];
#pragma unroll
    for (int v = 0; v < 8; ++v) { rmax[v] = -3.0e38f; rsum[v] = 0.0f; }

    const int kend = q0 + 16;                  // causal horizon (exclusive)
    for (int kb = 0; kb < kend; kb += 32) {
        const int key0 = kb + lo;
        const int key1 = key0 + 16;

        if (kb + 32 < kend)
            __builtin_prefetch(Kh + (size_t)(kb + 32 + lane) * HEAD_DIM, 0, 3);

        v8f s0, s1;
#pragma unroll
        for (int j = 0; j < 8; ++j) { s0[j] = 0.0f; s1[j] = 0.0f; }
#pragma unroll
        for (int c = 0; c < 4; ++c) {
            Frag b0, b1;
            const unsigned short* kp0 = Kh + (size_t)key0 * HEAD_DIM + c * 32 + hi * 8;
            const unsigned short* kp1 = Kh + (size_t)key1 * HEAD_DIM + c * 32 + hi * 8;
            b0.q[0] = *(const uint4*)kp0;  b0.q[1] = *(const uint4*)(kp0 + 16);
            b1.q[0] = *(const uint4*)kp1;  b1.q[1] = *(const uint4*)(kp1 + 16);
            s0 = __builtin_amdgcn_wmma_f32_16x16x32_bf16(
                false, qa[c].v, false, b0.v, (short)0, s0, false, false);
            s1 = __builtin_amdgcn_wmma_f32_16x16x32_bf16(
                false, qa[c].v, false, b1.v, (short)0, s1, false, false);
        }

        const float am0 = amask[key0];
        const float am1 = amask[key1 < S_LEN ? key1 : S_LEN - 1];

        float f0[8], f1[8], p0[8], p1[8];
#pragma unroll
        for (int v = 0; v < 8; ++v) {
            int row = q0 + v + 8 * hi;
            f0[v] = (key0 > row || am0 <= 0.0f) ? -1.0e30f : s0[v];
            f1[v] = (key1 > row || am1 <= 0.0f) ? -1.0e30f : s1[v];
        }
#pragma unroll
        for (int v = 0; v < 8; ++v) {
            float mx = fmaxf(f0[v], f1[v]);
#pragma unroll
            for (int off = 1; off < 16; off <<= 1)
                mx = fmaxf(mx, __shfl_xor(mx, off, 32));
            float nm = fmaxf(rmax[v], mx);
            float al = __expf(rmax[v] - nm);
            rmax[v] = nm;
            p0[v] = __expf(f0[v] - nm);
            p1[v] = __expf(f1[v] - nm);
            float ps = p0[v] + p1[v];
#pragma unroll
            for (int off = 1; off < 16; off <<= 1)
                ps += __shfl_xor(ps, off, 32);
            rsum[v] = rsum[v] * al + ps;
#pragma unroll
            for (int nt = 0; nt < 8; ++nt) O[nt][v] *= al;
        }

        __syncthreads();
#pragma unroll
        for (int v = 0; v < 8; ++v) {
            Pl[v + 8 * hi][lo]      = f2bf(p0[v]);
            Pl[v + 8 * hi][16 + lo] = f2bf(p1[v]);
        }
        __syncthreads();

        Frag pf;
        const unsigned short* pp = &Pl[lo][hi * 8];
        pf.q[0] = *(const uint4*)pp;
        pf.q[1] = *(const uint4*)(pp + 16);

#pragma unroll
        for (int nt = 0; nt < 8; ++nt) {
            Frag vf;
            const unsigned short* vp =
                Vh + (size_t)(nt * 16 + lo) * S_LEN + kb + hi * 8;
            vf.q[0] = *(const uint4*)vp;
            vf.q[1] = *(const uint4*)(vp + 16);
            O[nt] = __builtin_amdgcn_wmma_f32_16x16x32_bf16(
                false, pf.v, false, vf.v, (short)0, O[nt], false, false);
        }
        __syncthreads();
    }

#pragma unroll
    for (int nt = 0; nt < 8; ++nt)
#pragma unroll
        for (int v = 0; v < 8; ++v) {
            int row = q0 + v + 8 * hi;
            float denom = rsum[v];
            float val = O[nt][v] / (denom > 0.0f ? denom : 1.0f);
            ctx[(size_t)row * (N_HEADS * HEAD_DIM) + h * HEAD_DIM + nt * 16 + lo] =
                f2bf(val);
        }
}

// ---------------------------------------------------------------------------
// Host-side orchestration
// ---------------------------------------------------------------------------
extern "C" void kernel_launch(void* const* d_in, const int* in_sizes, int n_in,
                              void* d_out, int out_size, void* d_ws, size_t ws_size,
                              hipStream_t stream) {
    const float* hidden = (const float*)d_in[0];
    const float* amask  = (const float*)d_in[1];
    const int*   pos    = (const int*)d_in[2];
    const float* wq     = (const float*)d_in[3];
    const float* wk     = (const float*)d_in[4];
    const float* wv     = (const float*)d_in[5];
    const float* wo     = (const float*)d_in[6];
    float* out = (float*)d_out;

    size_t off = 0;
    auto alloc = [&](size_t bytes) -> void* {
        void* p = (char*)d_ws + off;
        off += (bytes + 255) & ~(size_t)255;
        return p;
    };

    unsigned short* Xb  = (unsigned short*)alloc((size_t)S_LEN * D_MODEL * 2);
    unsigned short* WqT = (unsigned short*)alloc((size_t)D_MODEL * D_MODEL * 2);
    unsigned short* WkT = (unsigned short*)alloc((size_t)D_MODEL * 1024 * 2);
    unsigned short* WvT = (unsigned short*)alloc((size_t)D_MODEL * 1024 * 2);
    unsigned short* WoT = (unsigned short*)alloc((size_t)D_MODEL * D_MODEL * 2);
    float* Qtmp = (float*)alloc((size_t)S_LEN * D_MODEL * 4);
    float* Ktmp = (float*)alloc((size_t)S_LEN * 1024 * 4);
    float* Vtmp = (float*)alloc((size_t)S_LEN * 1024 * 4);
    unsigned short* Qr = (unsigned short*)alloc((size_t)N_HEADS * S_LEN * HEAD_DIM * 2);
    unsigned short* Kr = (unsigned short*)alloc((size_t)N_KVH  * S_LEN * HEAD_DIM * 2);
    unsigned short* Vr = (unsigned short*)alloc((size_t)N_KVH  * HEAD_DIM * S_LEN * 2);
    unsigned short* Ctx = (unsigned short*)alloc((size_t)S_LEN * D_MODEL * 2);

    // 1) hidden -> bf16 row-major; weights -> bf16 TRANSPOSED (N x K)
    {
        int n = S_LEN * D_MODEL;
        int blocks = (n + 255) / 256;
        if (blocks > 8192) blocks = 8192;
        k_cvt_bf16<<<blocks, 256, 0, stream>>>(hidden, Xb, n);
    }
    k_cvt_tr<<<dim3(D_MODEL / 32, D_MODEL / 32), 256, 0, stream>>>(wq, WqT, D_MODEL, D_MODEL);
    k_cvt_tr<<<dim3(1024 / 32,    D_MODEL / 32), 256, 0, stream>>>(wk, WkT, D_MODEL, 1024);
    k_cvt_tr<<<dim3(1024 / 32,    D_MODEL / 32), 256, 0, stream>>>(wv, WvT, D_MODEL, 1024);
    k_cvt_tr<<<dim3(D_MODEL / 32, D_MODEL / 32), 256, 0, stream>>>(wo, WoT, D_MODEL, D_MODEL);

    // 2) Q/K/V projections (bf16 WMMA, fp32 out)
    k_gemm_bf16<<<dim3(D_MODEL / 128, S_LEN / 128), 256, 0, stream>>>(
        Xb, WqT, Qtmp, S_LEN, D_MODEL, D_MODEL);
    k_gemm_bf16<<<dim3(1024 / 128, S_LEN / 128), 256, 0, stream>>>(
        Xb, WkT, Ktmp, S_LEN, 1024, D_MODEL);
    k_gemm_bf16<<<dim3(1024 / 128, S_LEN / 128), 256, 0, stream>>>(
        Xb, WvT, Vtmp, S_LEN, 1024, D_MODEL);

    // 3) RoPE + repack (scale 1/sqrt(128) folded into Q)
    const float qscale = 1.0f / sqrtf((float)HEAD_DIM);
    {
        int n = S_LEN * N_HEADS * 64;
        k_rope_pack<<<(n + 255) / 256, 256, 0, stream>>>(Qtmp, pos, Qr, N_HEADS, qscale);
    }
    {
        int n = S_LEN * N_KVH * 64;
        k_rope_pack<<<(n + 255) / 256, 256, 0, stream>>>(Ktmp, pos, Kr, N_KVH, 1.0f);
    }
    {
        int n = S_LEN * N_KVH * HEAD_DIM;
        k_pack_vT<<<(n + 255) / 256, 256, 0, stream>>>(Vtmp, Vr);
    }

    // 4) flash attention
    k_attn<<<dim3(S_LEN / 16, N_HEADS), 32, 0, stream>>>(Qr, Kr, Vr, amask, Ctx);

    // 5) output projection -> fp32 d_out
    k_gemm_bf16<<<dim3(D_MODEL / 128, S_LEN / 128), 256, 0, stream>>>(
        Ctx, WoT, out, S_LEN, D_MODEL, D_MODEL);
}
